// SelfAttention_20512763805945
// MI455X (gfx1250) — matrix operand order repaired and checked by reading, not compiled, not run on hardware
//
#include <hip/hip_runtime.h>
#include <hip/hip_bf16.h>
#include <cstdint>

// Self-attention B=8, S=2048, D=1024. f16 WMMA (f32 accumulate), async global->LDS staging,
// double-buffered 64x64x64 tiles, fragment-preloaded inner loop. Workspace: ~295 MB.

typedef __attribute__((ext_vector_type(16))) _Float16 v16h;
typedef __attribute__((ext_vector_type(8)))  _Float16 v8h;
typedef __attribute__((ext_vector_type(4)))  _Float16 v4h;
typedef __attribute__((ext_vector_type(8)))  float    v8f;

#define B_  8
#define S_  2048
#define D_  1024
#define NR_ (B_ * S_)      // 16384 total rows
#define KSPLIT 4
#define TSTR 72            // halves per LDS tile row (64 data + 8 pad) = 144B, 36-bank stride

static __device__ __forceinline__ v8f wmma_f16(v16h a, v16h b, v8f c) {
    return __builtin_amdgcn_wmma_f32_16x16x32_f16(false, a, false, b, (short)0, c, false, false);
}

// ---- CDNA5 async global->LDS copy (16B per lane), tracked by ASYNCcnt ----
static __device__ __forceinline__ void cp16b_async(_Float16* lds, const _Float16* g) {
    unsigned l = (unsigned)(uintptr_t)lds;                 // low 32 bits of generic ptr = LDS offset
    unsigned long long ga = (unsigned long long)(uintptr_t)g;
    asm volatile("global_load_async_to_lds_b128 %0, %1, off"
                 :: "v"(l), "v"(ga) : "memory");
}
static __device__ __forceinline__ void async_wait0() {
    asm volatile("s_wait_asynccnt 0x0" ::: "memory");
}

// Stage a 64x64 f16 tile into LDS (stride TSTR) with 128 threads, 4 async b128 per thread.
static __device__ __forceinline__ void stage_async(_Float16* dst, const _Float16* src, int ld, int t) {
#pragma unroll
    for (int i = 0; i < 4; ++i) {
        int idx = (i * 128 + t) * 8;     // halves
        int r = idx >> 6, c = idx & 63;
        cp16b_async(dst + r * TSTR + c, src + (size_t)r * ld + c);
    }
}

// A-fragment 16x32 f16: lane l -> row m=l&15, halves [kg*8..+8 | 16+kg*8..+8], kg=l>>4.
static __device__ __forceinline__ v16h fragA(const _Float16* tile, int row0, int col0, int lane) {
    const _Float16* p = tile + (row0 + (lane & 15)) * TSTR + col0 + (lane >> 4) * 8;
    v8h lo = *(const v8h*)p;
    v8h hi = *(const v8h*)(p + 16);
    return __builtin_shufflevector(lo, hi, 0,1,2,3,4,5,6,7,8,9,10,11,12,13,14,15);
}
// B-fragment 32x16 f16 from column-major tile rows: lane l -> col n=l&15, halves [kg*8..+8 | 16+kg*8..+8], kg=l>>4.
static __device__ __forceinline__ v16h fragB(const _Float16* tile, int row0, int col0, int lane) {
    const _Float16* p = tile + (row0 + (lane & 15)) * TSTR + col0 + (lane >> 4) * 8;
    v8h lo = *(const v8h*)p;
    v8h hi = *(const v8h*)(p + 16);
    return __builtin_shufflevector(lo, hi, 0,1,2,3,4,5,6,7,8,9,10,11,12,13,14,15);
}

// 64-deep K-chunk: preload ALL fragments (20 ds_load_b128), then 8 wmma back-to-back.
static __device__ __forceinline__ void mma_stage(v8f acc[4], const _Float16* As,
                                                 const _Float16* Bs, int wave, int lane) {
    v16h a0 = fragA(As, wave * 16, 0,  lane);
    v16h a1 = fragA(As, wave * 16, 32, lane);
    v16h b0[4], b1[4];
#pragma unroll
    for (int nt = 0; nt < 4; ++nt) {
        b0[nt] = fragB(Bs, nt * 16, 0,  lane);
        b1[nt] = fragB(Bs, nt * 16, 32, lane);
    }
#pragma unroll
    for (int nt = 0; nt < 4; ++nt) acc[nt] = wmma_f16(a0, b0[nt], acc[nt]);
#pragma unroll
    for (int nt = 0; nt < 4; ++nt) acc[nt] = wmma_f16(a1, b1[nt], acc[nt]);
}

// Branch-free double-buffered GEMM core over K (steady state: prefetch, compute, wait+barrier).
// Kdim must be a multiple of 64 and >= 64.
static __device__ __forceinline__ void gemm_core(v8f acc[4], _Float16 As[2][64 * TSTR],
                                                 _Float16 Bs[2][64 * TSTR],
                                                 const _Float16* srcA, const _Float16* srcB,
                                                 int ldA, int ldB, int Kdim,
                                                 int t, int wave, int lane) {
    stage_async(As[0], srcA, ldA, t);
    stage_async(Bs[0], srcB, ldB, t);
    async_wait0(); __syncthreads();
    int buf = 0;
    for (int k0 = 0; k0 < Kdim - 64; k0 += 64, buf ^= 1) {
        stage_async(As[buf ^ 1], srcA + k0 + 64, ldA, t);
        stage_async(Bs[buf ^ 1], srcB + k0 + 64, ldB, t);
        mma_stage(acc, As[buf], Bs[buf], wave, lane);
        async_wait0(); __syncthreads();
    }
    mma_stage(acc, As[buf], Bs[buf], wave, lane);   // peeled last chunk, no prefetch/branch
}

// ---------------- Kernel: f32 -> f16 elementwise ----------------
__global__ __launch_bounds__(256)
void cvt_f16_kernel(const float* __restrict__ s, _Float16* __restrict__ d, int n) {
    int i = (blockIdx.x * 256 + threadIdx.x) * 4;
    if (i + 3 < n) {
        float4 v = *(const float4*)(s + i);
        v4h h = {(_Float16)v.x, (_Float16)v.y, (_Float16)v.z, (_Float16)v.w};
        *(v4h*)(d + i) = h;
    }
}

// ---------------- Kernel: projection GEMM  Y = X*W^T + b  (all-f16 async tiles) ----------------
template <int TRANSPOSE_V>
__global__ __launch_bounds__(128)
void proj_kernel(const _Float16* __restrict__ Xh, const _Float16* __restrict__ Wh,
                 const float* __restrict__ bias, _Float16* __restrict__ Out) {
    __shared__ _Float16 As[2][64 * TSTR];
    __shared__ _Float16 Bs[2][64 * TSTR];
    const int t = threadIdx.x, lane = t & 31, wave = t >> 5;
    const int nBase = blockIdx.x * 64;
    const int mBase = blockIdx.y * 64;

    v8f acc[4] = {};
    gemm_core(acc, As, Bs, Xh + (size_t)mBase * D_, Wh + (size_t)nBase * D_,
              D_, D_, D_, t, wave, lane);

    const int nCol = lane & 15, grp = lane >> 4;
#pragma unroll
    for (int nt = 0; nt < 4; ++nt) {
        int nG = nBase + nt * 16 + nCol;
        float bv = bias[nG];
#pragma unroll
        for (int j = 0; j < 8; ++j) {
            int mG = mBase + wave * 16 + grp * 8 + j;
            float v = acc[nt][j] + bv;
            if (TRANSPOSE_V) {       // Vt[b][d][s]
                int bb = mG >> 11, ss = mG & (S_ - 1);
                Out[((size_t)bb * D_ + nG) * S_ + ss] = (_Float16)v;
            } else {
                Out[(size_t)mG * D_ + nG] = (_Float16)v;
            }
        }
    }
}

// ---------------- Kernel: scores = mask(Q K^T / 32) + partial online row stats ----------------
__global__ __launch_bounds__(128)
void scores_kernel(const _Float16* __restrict__ Qh, const _Float16* __restrict__ Kh,
                   const int* __restrict__ mask, float* __restrict__ Sc,
                   float* __restrict__ mPart, float* __restrict__ lPart) {
    __shared__ _Float16 As[2][64 * TSTR];
    __shared__ _Float16 Bs[2][64 * TSTR];
    const int t = threadIdx.x, lane = t & 31, wave = t >> 5;
    const size_t rowBase = (size_t)blockIdx.x * 64;     // global q row (b*S+q)
    const int split = blockIdx.y;                       // 0..KSPLIT-1
    const int b = (int)(rowBase >> 11);
    const _Float16* Qb = Qh + rowBase * D_;
    const _Float16* Kb = Kh + (size_t)b * S_ * D_;
    const int nCol = lane & 15, grp = lane >> 4;
    const float scale = 0.03125f;                       // 1/sqrt(1024)
    const int kBeg = split * (S_ / KSPLIT), kEnd = kBeg + S_ / KSPLIT;

    float m_run[8], l_run[8];
#pragma unroll
    for (int j = 0; j < 8; ++j) { m_run[j] = -3.0e38f; l_run[j] = 0.f; }

    for (int kT = kBeg; kT < kEnd; kT += 64) {
        v8f acc[4] = {};
        gemm_core(acc, As, Bs, Qb, Kb + (size_t)kT * D_, D_, D_, D_, t, wave, lane);

        float sv[4][8];
#pragma unroll
        for (int nt = 0; nt < 4; ++nt) {
            int kG = kT + nt * 16 + nCol;
            int mk = mask[(size_t)b * S_ + kG];
#pragma unroll
            for (int j = 0; j < 8; ++j)
                sv[nt][j] = mk ? acc[nt][j] * scale : -1.0e30f;
        }
#pragma unroll
        for (int nt = 0; nt < 4; ++nt) {
            int kG = kT + nt * 16 + nCol;
#pragma unroll
            for (int j = 0; j < 8; ++j) {
                int mG = wave * 16 + grp * 8 + j;
                Sc[(rowBase + mG) * S_ + kG] = sv[nt][j];
            }
        }
#pragma unroll
        for (int j = 0; j < 8; ++j) {
            float mx = fmaxf(fmaxf(sv[0][j], sv[1][j]), fmaxf(sv[2][j], sv[3][j]));
#pragma unroll
            for (int off = 1; off < 16; off <<= 1)
                mx = fmaxf(mx, __shfl_xor(mx, off));
            float m_new = fmaxf(m_run[j], mx);
            float p = __expf(sv[0][j] - m_new) + __expf(sv[1][j] - m_new) +
                      __expf(sv[2][j] - m_new) + __expf(sv[3][j] - m_new);
#pragma unroll
            for (int off = 1; off < 16; off <<= 1)
                p += __shfl_xor(p, off);
            l_run[j] = l_run[j] * __expf(m_run[j] - m_new) + p;
            m_run[j] = m_new;
        }
        __syncthreads();   // ensure all waves done with peeled buffer before next tile restages
    }
    if (nCol == 0) {
#pragma unroll
        for (int j = 0; j < 8; ++j) {
            size_t r = rowBase + wave * 16 + grp * 8 + j;
            mPart[(size_t)split * NR_ + r] = m_run[j];
            lPart[(size_t)split * NR_ + r] = l_run[j];
        }
    }
}

// ---------------- Kernel: combine partial stats ----------------
__global__ __launch_bounds__(256)
void stats_kernel(const float* __restrict__ mPart, const float* __restrict__ lPart,
                  float* __restrict__ rowM, float* __restrict__ rowLinv) {
    int r = blockIdx.x * 256 + threadIdx.x;
    if (r >= NR_) return;
    float m = -3.0e38f;
#pragma unroll
    for (int s = 0; s < KSPLIT; ++s) m = fmaxf(m, mPart[(size_t)s * NR_ + r]);
    float l = 0.f;
#pragma unroll
    for (int s = 0; s < KSPLIT; ++s)
        l += lPart[(size_t)s * NR_ + r] * __expf(mPart[(size_t)s * NR_ + r] - m);
    rowM[r] = m;
    rowLinv[r] = 1.0f / l;
}

// ---------------- Kernel: P = exp(s - m) / l  -> f16 ----------------
__global__ __launch_bounds__(256)
void pexp_kernel(const float* __restrict__ Sc, const float* __restrict__ rowM,
                 const float* __restrict__ rowLinv, _Float16* __restrict__ Ph) {
    size_t i = ((size_t)blockIdx.x * 256 + threadIdx.x) * 4;
    size_t r = i >> 11;                                  // / S_
    float m = rowM[r], li = rowLinv[r];
    float4 s4 = *(const float4*)(Sc + i);
    v4h h = {(_Float16)(__expf(s4.x - m) * li), (_Float16)(__expf(s4.y - m) * li),
             (_Float16)(__expf(s4.z - m) * li), (_Float16)(__expf(s4.w - m) * li)};
    *(v4h*)(Ph + i) = h;
}

// ---------------- Kernel: out = P * V  (all-f16 async tiles, f32 out) ----------------
__global__ __launch_bounds__(128)
void out_kernel(const _Float16* __restrict__ Ph, const _Float16* __restrict__ Vt,
                float* __restrict__ Out) {
    __shared__ _Float16 As[2][64 * TSTR];
    __shared__ _Float16 Bs[2][64 * TSTR];
    const int t = threadIdx.x, lane = t & 31, wave = t >> 5;
    const int nBase = blockIdx.x * 64;                  // dv
    const size_t rowBase = (size_t)blockIdx.y * 64;     // global q row
    const int b = (int)(rowBase >> 11);

    v8f acc[4] = {};
    gemm_core(acc, As, Bs, Ph + rowBase * S_,
              Vt + (size_t)b * D_ * S_ + (size_t)nBase * S_,
              S_, S_, S_, t, wave, lane);

    const int nCol = lane & 15, grp = lane >> 4;
#pragma unroll
    for (int nt = 0; nt < 4; ++nt) {
        int nG = nBase + nt * 16 + nCol;
#pragma unroll
        for (int j = 0; j < 8; ++j) {
            size_t mG = rowBase + wave * 16 + grp * 8 + j;
            Out[mG * D_ + nG] = acc[nt][j];
        }
    }
}

extern "C" void kernel_launch(void* const* d_in, const int* in_sizes, int n_in,
                              void* d_out, int out_size, void* d_ws, size_t ws_size,
                              hipStream_t stream) {
    const float* q  = (const float*)d_in[0];
    const float* k  = (const float*)d_in[1];
    const float* v  = (const float*)d_in[2];
    const int*   mk = (const int*)  d_in[3];
    const float* Wq = (const float*)d_in[4];
    const float* bq = (const float*)d_in[5];
    const float* Wk = (const float*)d_in[6];
    const float* bk = (const float*)d_in[7];
    const float* Wv = (const float*)d_in[8];
    const float* bv = (const float*)d_in[9];
    float* out = (float*)d_out;

    char* ws = (char*)d_ws;
    const size_t szXh  = (size_t)NR_ * D_ * sizeof(_Float16);     // 32 MB
    const size_t szW   = (size_t)D_ * D_ * sizeof(_Float16);      // 2 MB
    const size_t szSc  = (size_t)NR_ * S_ * sizeof(float);        // 128 MB
    const size_t szRow = (size_t)NR_ * sizeof(float);             // 64 KB
    size_t off = 0;
    _Float16* Qh  = (_Float16*)(ws + off); off += szXh;
    _Float16* Kh  = (_Float16*)(ws + off); off += szXh;
    _Float16* Vt  = (_Float16*)(ws + off); off += szXh;
    _Float16* Wqh = (_Float16*)(ws + off); off += szW;
    _Float16* Wkh = (_Float16*)(ws + off); off += szW;
    _Float16* Wvh = (_Float16*)(ws + off); off += szW;
    float* rM     = (float*)(ws + off); off += szRow;
    float* rL     = (float*)(ws + off); off += szRow;
    float* mPart  = (float*)(ws + off); off += szRow * KSPLIT;
    float* lPart  = (float*)(ws + off); off += szRow * KSPLIT;
    // region2: xh (phase 1) overlapped by Sc/Ph (phase 2)
    char* region2 = ws + off;
    _Float16* xqh = (_Float16*)(region2);
    _Float16* xkh = (_Float16*)(region2 + szXh);
    _Float16* xvh = (_Float16*)(region2 + 2 * szXh);
    float*    Sc  = (float*)   (region2);
    _Float16* Ph  = (_Float16*)(region2 + szSc);

    const int nX = NR_ * D_, nW = D_ * D_;
    cvt_f16_kernel<<<nX / 1024, 256, 0, stream>>>(q,  xqh, nX);
    cvt_f16_kernel<<<nX / 1024, 256, 0, stream>>>(k,  xkh, nX);
    cvt_f16_kernel<<<nX / 1024, 256, 0, stream>>>(v,  xvh, nX);
    cvt_f16_kernel<<<nW / 1024, 256, 0, stream>>>(Wq, Wqh, nW);
    cvt_f16_kernel<<<nW / 1024, 256, 0, stream>>>(Wk, Wkh, nW);
    cvt_f16_kernel<<<nW / 1024, 256, 0, stream>>>(Wv, Wvh, nW);

    dim3 gProj(D_ / 64, NR_ / 64);                                 // (16, 256)
    proj_kernel<0><<<gProj, 128, 0, stream>>>(xqh, Wqh, bq, Qh);
    proj_kernel<0><<<gProj, 128, 0, stream>>>(xkh, Wkh, bk, Kh);
    proj_kernel<1><<<gProj, 128, 0, stream>>>(xvh, Wvh, bv, Vt);

    dim3 gSc(NR_ / 64, KSPLIT);                                    // (256, 4)
    scores_kernel<<<gSc, 128, 0, stream>>>(Qh, Kh, mk, Sc, mPart, lPart);
    stats_kernel<<<NR_ / 256, 256, 0, stream>>>(mPart, lPart, rM, rL);
    pexp_kernel<<<(NR_ * (S_ / 4)) / 256, 256, 0, stream>>>(Sc, rM, rL, Ph);

    dim3 gOut(D_ / 64, NR_ / 64);                                  // (16, 256)
    out_kernel<<<gOut, 128, 0, stream>>>(Ph, Vt, out);
}
